// TGNN_26070451487092
// MI455X (gfx1250) — matrix operand order repaired and checked by reading, not compiled, not run on hardware
//
#include <hip/hip_runtime.h>
#include <hip/hip_bf16.h>

// ---------------------------------------------------------------------------
// GAT forward for MI455X (gfx1250, wave32).
//   - GEMMs via v_wmma_f32_16x16x32_bf16, 128x128x32 tiles, 8 waves/block.
//   - bf16 operands in global ws (A converted, B transposed): halves HBM
//     traffic; tiles are contiguous b128 copies.
//   - Double-buffered GLOBAL_LOAD_ASYNC_TO_LDS_B128 pipeline: next tile's
//     loads overlap current tile's WMMAs; s_wait_asynccnt(4) drains exactly
//     the previous per-wave batch (4 async instructions per tile).
//   - LDS stride 40 halves (80B): every ISA fragment chunk is a 16B-aligned
//     contiguous ds_load_b128.
//   - Edge-encoder GEMM and a_e collapsed algebraically; segment softmax in
//     3 recompute passes with deferred normalization.
// ---------------------------------------------------------------------------

typedef __attribute__((ext_vector_type(16))) __bf16 v16bf;
typedef __attribute__((ext_vector_type(8)))  __bf16 v8bf;
typedef __attribute__((ext_vector_type(8)))  float  v8f;
typedef int v4i_async __attribute__((vector_size(16)));
typedef __attribute__((address_space(1))) v4i_async* g_v4i_p;
typedef __attribute__((address_space(3))) v4i_async* l_v4i_p;

static constexpr int N_     = 20000;
static constexpr int E_     = 200000;
static constexpr int ETOT_  = E_ + N_;
static constexpr int ND_    = 128;
static constexpr int ED_    = 64;
static constexpr int H_     = 256;
static constexpr int HEADS_ = 4;
static constexpr int L_     = 3;

#if defined(__gfx1250__) && __has_builtin(__builtin_amdgcn_global_load_async_to_lds_b128) && __has_builtin(__builtin_amdgcn_s_wait_asynccnt)
#define USE_ASYNC_LDS 1
#else
#define USE_ASYNC_LDS 0
#endif

// ------------------------------- helpers -----------------------------------
__device__ __forceinline__ __bf16 f2bf(float f) {
    union { unsigned short s; __bf16 b; } c;
    c.s = (unsigned short)(__float_as_uint(f) >> 16);
    return c.b;
}
__device__ __forceinline__ unsigned flipf(float f) {    // order-preserving f32 -> u32
    unsigned u = __float_as_uint(f);
    return (u & 0x80000000u) ? ~u : (u | 0x80000000u);
}
__device__ __forceinline__ float unflipf(unsigned u) {
    unsigned b = (u & 0x80000000u) ? (u ^ 0x80000000u) : ~u;
    return __uint_as_float(b);
}
__device__ __forceinline__ float wave_sum(float v) {    // wave32 butterfly
#pragma unroll
    for (int off = 16; off > 0; off >>= 1) v += __shfl_xor(v, off, 32);
    return v;
}
__device__ __forceinline__ void edge_sdw(int e, const int* __restrict__ src0,
                                         const int* __restrict__ dst0,
                                         const float* __restrict__ ew,
                                         const float* __restrict__ loopw,
                                         int& s, int& d, float& w) {
    if (e < E_) { s = src0[e]; d = dst0[e]; w = ew[e]; }
    else        { s = d = e - E_; w = loopw[e - E_]; }
}
__device__ __forceinline__ float edge_logit(int s, int d, float w, int h,
                                            const float* __restrict__ a_s,
                                            const float* __restrict__ a_d,
                                            const float* __restrict__ cvec) {
    float l = a_s[s * HEADS_ + h] + a_d[d * HEADS_ + h] + w * cvec[h];
    return l > 0.f ? l : 0.2f * l;
}

// ------------------------------- utility kernels ---------------------------
__global__ void k_fill(float* __restrict__ p, float v, int n) {
    int i = blockIdx.x * blockDim.x + threadIdx.x;
    if (i < n) p[i] = v;
}
__global__ void k_deg(const int* __restrict__ dst0, float* __restrict__ deg) {
    int e = blockIdx.x * blockDim.x + threadIdx.x;
    if (e < E_) atomicAdd(&deg[dst0[e]], 1.0f);
}
__global__ void k_cvt_bf16(const float* __restrict__ src, __bf16* __restrict__ dst, int n) {
    int i = blockIdx.x * blockDim.x + threadIdx.x;
    if (i < n) dst[i] = f2bf(src[i]);
}
// f32 [K,Nc] -> bf16 transposed [Nc,K]
__global__ void k_tr_bf16(const float* __restrict__ B, __bf16* __restrict__ Bt,
                          int K, int Nc) {
    int i = blockIdx.x * blockDim.x + threadIdx.x;
    if (i >= K * Nc) return;
    int k = i / Nc, c = i - k * Nc;
    Bt[(size_t)c * K + k] = f2bf(B[i]);
}
__global__ void k_copy(const float* __restrict__ src, float* __restrict__ dst, int n) {
    int i = blockIdx.x * blockDim.x + threadIdx.x;
    if (i < n) dst[i] = src[i];
}

// ----------------------- WMMA bf16 GEMM: C = act(A@B + bias) ---------------
// A  : bf16 [M,K] row-major          (K % 32 == 0, 16B-aligned rows)
// Bt : bf16 [Nc,K] (B transposed)    (Nc % 128 == 0)
#define BM 128
#define BN 128
#define BK 32
#define LDT 40   // LDS tile stride in halves (80 B): keeps frag chunks 16B-aligned

__device__ __forceinline__ void stage_chunk_sync(const __bf16* __restrict__ g,
                                                 __bf16* __restrict__ l, bool ok) {
    uint4 v = ok ? *(const uint4*)g : uint4{0u, 0u, 0u, 0u};
    *(uint4*)l = v;
}

__device__ __forceinline__ void stage_tiles_sync(const __bf16* __restrict__ A,
                                                 const __bf16* __restrict__ Bt,
                                                 int rowBase, int colBase, int K, int k0,
                                                 __bf16* __restrict__ As,
                                                 __bf16* __restrict__ Bs,
                                                 int tid, int M) {
    for (int i = tid; i < 512; i += 256) {
        int r = i >> 2, ch = i & 3;
        stage_chunk_sync(A + (size_t)(rowBase + r) * K + k0 + ch * 8,
                         &As[r * LDT + ch * 8], (rowBase + r) < M);
    }
    for (int i = tid; i < 512; i += 256) {
        int r = i >> 2, ch = i & 3;
        stage_chunk_sync(Bt + (size_t)(colBase + r) * K + k0 + ch * 8,
                         &Bs[r * LDT + ch * 8], true);
    }
}

#if USE_ASYNC_LDS
// 4 async instructions per wave per call (2 A-chunks + 2 B-chunks per thread)
__device__ __forceinline__ void stage_tiles_async(const __bf16* __restrict__ A,
                                                  const __bf16* __restrict__ Bt,
                                                  int rowBase, int colBase, int K, int k0,
                                                  __bf16* __restrict__ As,
                                                  __bf16* __restrict__ Bs,
                                                  int tid) {
    for (int i = tid; i < 512; i += 256) {
        int r = i >> 2, ch = i & 3;
        __builtin_amdgcn_global_load_async_to_lds_b128(
            (g_v4i_p)(A + (size_t)(rowBase + r) * K + k0 + ch * 8),
            (l_v4i_p)(&As[r * LDT + ch * 8]), 0, 0);
    }
    for (int i = tid; i < 512; i += 256) {
        int r = i >> 2, ch = i & 3;
        __builtin_amdgcn_global_load_async_to_lds_b128(
            (g_v4i_p)(Bt + (size_t)(colBase + r) * K + k0 + ch * 8),
            (l_v4i_p)(&Bs[r * LDT + ch * 8]), 0, 0);
    }
}
#endif

__device__ __forceinline__ void compute_tile(const __bf16* __restrict__ As,
                                             const __bf16* __restrict__ Bs,
                                             int wm, int wn, int frow, int kg,
                                             v8f acc[4][2]) {
    v16bf afr[4];
#pragma unroll
    for (int t = 0; t < 4; ++t) {
        const __bf16* ap = &As[(wm * 64 + t * 16 + frow) * LDT];
        v8bf lo = *(const v8bf*)(ap + kg * 8);        // K = kg*8 + 0..7
        v8bf hi = *(const v8bf*)(ap + 16 + kg * 8);   // K = 16 + kg*8 + 0..7
        afr[t] = __builtin_shufflevector(lo, hi, 0, 1, 2, 3, 4, 5, 6, 7,
                                         8, 9, 10, 11, 12, 13, 14, 15);
    }
    v16bf bfr[2];
#pragma unroll
    for (int u = 0; u < 2; ++u) {
        const __bf16* bp = &Bs[(wn * 32 + u * 16 + frow) * LDT + kg * 8];
        v8bf lo = *(const v8bf*)(bp);                 // K = kg*8 + 0..7
        v8bf hi = *(const v8bf*)(bp + 16);            // K = 16 + kg*8 + 0..7
        bfr[u] = __builtin_shufflevector(lo, hi, 0, 1, 2, 3, 4, 5, 6, 7,
                                         8, 9, 10, 11, 12, 13, 14, 15);
    }
#pragma unroll
    for (int t = 0; t < 4; ++t)
#pragma unroll
        for (int u = 0; u < 2; ++u)
            acc[t][u] = __builtin_amdgcn_wmma_f32_16x16x32_bf16(
                false, afr[t], false, bfr[u], (short)0, acc[t][u], false, false);
}

__global__ __launch_bounds__(256)
void k_gemm_wmma(const __bf16* __restrict__ A, const __bf16* __restrict__ Bt,
                 const float* __restrict__ bias, float* __restrict__ C,
                 int M, int K, int Nc, int do_relu) {
    __shared__ alignas(16) __bf16 As[2][BM * LDT];
    __shared__ alignas(16) __bf16 Bs[2][BN * LDT];

    const int tid  = threadIdx.x;
    const int lane = tid & 31;
    const int wave = tid >> 5;
    const int wm   = wave >> 2;      // 0..1 -> 64-row strip
    const int wn   = wave & 3;       // 0..3 -> 32-col strip
    const int rowBase = blockIdx.y * BM;
    const int colBase = blockIdx.x * BN;
    const bool fullTile = (rowBase + BM) <= M;
    const int frow = lane & 15;
    const int kg   = lane >> 4;
    const int nIter = K / BK;

    v8f acc[4][2];
#pragma unroll
    for (int t = 0; t < 4; ++t)
#pragma unroll
        for (int u = 0; u < 2; ++u)
#pragma unroll
            for (int i = 0; i < 8; ++i) acc[t][u][i] = 0.0f;

#if USE_ASYNC_LDS
    if (fullTile) {
        // double-buffered async pipeline: tile it+1 loads overlap tile it WMMAs
        stage_tiles_async(A, Bt, rowBase, colBase, K, 0, As[0], Bs[0], tid);
        for (int it = 0; it < nIter; ++it) {
            int cur = it & 1;
            if (it + 1 < nIter) {
                stage_tiles_async(A, Bt, rowBase, colBase, K, (it + 1) * BK,
                                  As[cur ^ 1], Bs[cur ^ 1], tid);
                __builtin_amdgcn_s_wait_asynccnt(4);   // drain previous batch only
            } else {
                __builtin_amdgcn_s_wait_asynccnt(0);
            }
            __syncthreads();                 // tile `cur` visible to all waves
            compute_tile(As[cur], Bs[cur], wm, wn, frow, kg, acc);
            __syncthreads();                 // reads done before buffer reuse
        }
    } else
#endif
    {
        for (int it = 0; it < nIter; ++it) {
            __syncthreads();
            stage_tiles_sync(A, Bt, rowBase, colBase, K, it * BK, As[0], Bs[0], tid, M);
            __syncthreads();
            compute_tile(As[0], Bs[0], wm, wn, frow, kg, acc);
        }
    }

    // ---- store: D rows = (lane>>4)*8 + r, col = lane&15 ----
#pragma unroll
    for (int t = 0; t < 4; ++t)
#pragma unroll
        for (int u = 0; u < 2; ++u) {
            int row0 = rowBase + wm * 64 + t * 16 + (lane >> 4) * 8;
            int col  = colBase + wn * 32 + u * 16 + (lane & 15);
#pragma unroll
            for (int r = 0; r < 8; ++r) {
                int row = row0 + r;
                if (row < M) {
                    float v = acc[t][u][r];
                    if (bias)    v += bias[col];
                    if (do_relu) v = fmaxf(v, 0.0f);
                    C[(size_t)row * Nc + col] = v;
                }
            }
        }
}

// ----------------------- per-layer small precomputes ------------------------
__global__ void k_prep_layer(const float* __restrict__ eew, const float* __restrict__ eeb,
                             const float* __restrict__ lew, const float* __restrict__ aet,
                             float* __restrict__ wmean, float* __restrict__ cvec,
                             float* __restrict__ bmean) {
    int t = threadIdx.x;
    if (t < ED_) {
        float s = 0.f;
        for (int j = 0; j < H_; ++j) s += eew[t * H_ + j];
        wmean[t] = s * (1.0f / H_);
    } else if (t == ED_) {
        float s = 0.f;
        for (int j = 0; j < H_; ++j) s += eeb[j];
        *bmean = s * (1.0f / H_);
    } else if (t < ED_ + 1 + HEADS_) {
        int h = t - ED_ - 1;
        float s = 0.f;
        for (int d = 0; d < H_; ++d) s += lew[h * H_ + d] * aet[h * H_ + d];
        cvec[h] = s;
    }
}

__global__ void k_ew(const float* __restrict__ ea, const int* __restrict__ dst0,
                     const float* __restrict__ wmean, const float* __restrict__ bmean,
                     float* __restrict__ ew, float* __restrict__ loopw) {
    int e = blockIdx.x * blockDim.x + threadIdx.x;
    if (e >= E_) return;
    float acc = 0.f;
    const float* row = ea + (size_t)e * ED_;
#pragma unroll 8
    for (int d = 0; d < ED_; ++d) acc += row[d] * wmean[d];
    float v = acc + *bmean;
    ew[e] = v;
    atomicAdd(&loopw[dst0[e]], v);
}

__global__ void k_loopw_div(float* __restrict__ loopw, const float* __restrict__ deg) {
    int n = blockIdx.x * blockDim.x + threadIdx.x;
    if (n < N_) loopw[n] /= fmaxf(deg[n], 1.0f);
}

__global__ __launch_bounds__(256)
void k_asd(const float* __restrict__ xp, const float* __restrict__ as_w,
           const float* __restrict__ ad_w, float* __restrict__ a_s,
           float* __restrict__ a_d) {
    int gw = (blockIdx.x << 3) + (threadIdx.x >> 5);
    if (gw >= N_ * HEADS_) return;
    int n = gw >> 2, h = gw & 3, lane = threadIdx.x & 31;
    float ss = 0.f, sd = 0.f;
#pragma unroll
    for (int i = 0; i < 8; ++i) {
        int c = lane + i * 32;
        float xv = xp[(size_t)n * (HEADS_ * H_) + h * H_ + c];
        ss += xv * as_w[h * H_ + c];
        sd += xv * ad_w[h * H_ + c];
    }
    ss = wave_sum(ss);
    sd = wave_sum(sd);
    if (lane == 0) { a_s[n * HEADS_ + h] = ss; a_d[n * HEADS_ + h] = sd; }
}

__global__ void k_logit_max(const int* __restrict__ src0, const int* __restrict__ dst0,
                            const float* __restrict__ ew, const float* __restrict__ loopw,
                            const float* __restrict__ a_s, const float* __restrict__ a_d,
                            const float* __restrict__ cvec, unsigned* __restrict__ mU) {
    int e = blockIdx.x * blockDim.x + threadIdx.x;
    if (e >= ETOT_) return;
    int s, d; float w;
    edge_sdw(e, src0, dst0, ew, loopw, s, d, w);
#pragma unroll
    for (int h = 0; h < HEADS_; ++h) {
        float l = edge_logit(s, d, w, h, a_s, a_d, cvec);
        atomicMax(&mU[d * HEADS_ + h], flipf(l));
    }
}

__global__ void k_p_sum(const int* __restrict__ src0, const int* __restrict__ dst0,
                        const float* __restrict__ ew, const float* __restrict__ loopw,
                        const float* __restrict__ a_s, const float* __restrict__ a_d,
                        const float* __restrict__ cvec, const unsigned* __restrict__ mU,
                        float* __restrict__ ssum) {
    int e = blockIdx.x * blockDim.x + threadIdx.x;
    if (e >= ETOT_) return;
    int s, d; float w;
    edge_sdw(e, src0, dst0, ew, loopw, s, d, w);
#pragma unroll
    for (int h = 0; h < HEADS_; ++h) {
        float l = edge_logit(s, d, w, h, a_s, a_d, cvec);
        float p = __expf(l - unflipf(mU[d * HEADS_ + h]));
        atomicAdd(&ssum[d * HEADS_ + h], p);
    }
}

__global__ __launch_bounds__(256)
void k_scatter(const int* __restrict__ src0, const int* __restrict__ dst0,
               const float* __restrict__ ew, const float* __restrict__ loopw,
               const float* __restrict__ a_s, const float* __restrict__ a_d,
               const float* __restrict__ cvec, const unsigned* __restrict__ mU,
               const float* __restrict__ ssum, const float* __restrict__ xp,
               float* __restrict__ hbuf) {
    int e = (blockIdx.x << 3) + (threadIdx.x >> 5);
    if (e >= ETOT_) return;
    int lane = threadIdx.x & 31;
    int s, d; float w;
    edge_sdw(e, src0, dst0, ew, loopw, s, d, w);
    float alpha[HEADS_];
#pragma unroll
    for (int h = 0; h < HEADS_; ++h) {
        float l = edge_logit(s, d, w, h, a_s, a_d, cvec);
        float p = __expf(l - unflipf(mU[d * HEADS_ + h]));
        alpha[h] = 0.25f * p / ssum[d * HEADS_ + h];
    }
    const float* xrow = xp + (size_t)s * (HEADS_ * H_);
#pragma unroll
    for (int i = 0; i < 8; ++i) {
        int j = lane + i * 32;
        float v = alpha[0] * xrow[j] + alpha[1] * xrow[H_ + j] +
                  alpha[2] * xrow[2 * H_ + j] + alpha[3] * xrow[3 * H_ + j];
        atomicAdd(&hbuf[(size_t)d * H_ + j], v);
    }
}

__global__ __launch_bounds__(256)
void k_post(float* __restrict__ x, const float* __restrict__ hbuf,
            const float* __restrict__ gb, const float* __restrict__ g,
            const float* __restrict__ b) {
    int n = (blockIdx.x << 3) + (threadIdx.x >> 5);
    if (n >= N_) return;
    int lane = threadIdx.x & 31;
    float v[8];
    float s = 0.f;
#pragma unroll
    for (int i = 0; i < 8; ++i) {
        int c = lane + i * 32;
        v[i] = hbuf[(size_t)n * H_ + c] + gb[c];
        s += v[i];
    }
    float mu = wave_sum(s) * (1.0f / H_);
    float vs = 0.f;
#pragma unroll
    for (int i = 0; i < 8; ++i) { float dlt = v[i] - mu; vs += dlt * dlt; }
    float var = wave_sum(vs) * (1.0f / H_);
    float inv = rsqrtf(var + 1e-5f);
#pragma unroll
    for (int i = 0; i < 8; ++i) {
        int c = lane + i * 32;
        float hn = g[c] * (v[i] - mu) * inv + b[c];
        x[(size_t)n * H_ + c] = fmaxf(x[(size_t)n * H_ + c] + hn, 0.0f);
    }
}

__global__ __launch_bounds__(128)
void k_heads(const float* __restrict__ emb, const float* __restrict__ w1,
             const float* __restrict__ b1, const float* __restrict__ w2,
             const float* __restrict__ b2, float* __restrict__ out) {
    __shared__ float se[H_];
    __shared__ float red[4];
    int blk = blockIdx.x;
    int n = blk >> 2, k = blk & 3;
    int t = threadIdx.x;
    se[t] = emb[(size_t)n * H_ + t];
    se[t + 128] = emb[(size_t)n * H_ + t + 128];
    __syncthreads();
    float acc = b1[k * 128 + t];
    const float* w1p = w1 + (size_t)k * H_ * 128 + t;
    for (int d = 0; d < H_; ++d) acc += se[d] * w1p[(size_t)d * 128];
    acc = fmaxf(acc, 0.0f) * w2[k * 128 + t];
#pragma unroll
    for (int off = 16; off > 0; off >>= 1) acc += __shfl_down(acc, off, 32);
    if ((t & 31) == 0) red[t >> 5] = acc;
    __syncthreads();
    if (t == 0) {
        float o = red[0] + red[1] + red[2] + red[3] + b2[k];
        if (k == 0 || k == 3) o = 1.0f / (1.0f + __expf(-o));
        out[k * N_ + n] = o;
    }
}

// --------------------------------- launch ----------------------------------
extern "C" void kernel_launch(void* const* d_in, const int* in_sizes, int n_in,
                              void* d_out, int out_size, void* d_ws, size_t ws_size,
                              hipStream_t stream) {
    (void)in_sizes; (void)n_in; (void)out_size; (void)ws_size;

    const float* node_features = (const float*)d_in[0];
    const float* edge_attr     = (const float*)d_in[1];
    const float* enc_w         = (const float*)d_in[2];
    const float* enc_b         = (const float*)d_in[3];
    const float* edge_enc_w    = (const float*)d_in[4];
    const float* edge_enc_b    = (const float*)d_in[5];
    const float* gat_w         = (const float*)d_in[6];
    const float* att_src       = (const float*)d_in[7];
    const float* att_dst       = (const float*)d_in[8];
    const float* att_edge      = (const float*)d_in[9];
    const float* lin_edge_w    = (const float*)d_in[10];
    const float* gat_b         = (const float*)d_in[11];
    const float* ln_g          = (const float*)d_in[12];
    const float* ln_b          = (const float*)d_in[13];
    const float* head_w1       = (const float*)d_in[14];
    const float* head_b1       = (const float*)d_in[15];
    const float* head_w2       = (const float*)d_in[16];
    const float* head_b2       = (const float*)d_in[17];
    const int*   edge_index    = (const int*)d_in[18];
    const int* src0 = edge_index;
    const int* dst0 = edge_index + E_;
    float* out = (float*)d_out;

    // ----- workspace layout (floats, then bf16 tail) -----
    float* ws   = (float*)d_ws;
    float* x    = ws;                              // N*H
    float* hbuf = x    + (size_t)N_ * H_;          // N*H
    float* xp   = hbuf + (size_t)N_ * H_;          // N*4H
    float* a_s  = xp   + (size_t)N_ * HEADS_ * H_; // N*4
    float* a_d  = a_s  + (size_t)N_ * HEADS_;      // N*4
    float* m_f  = a_d  + (size_t)N_ * HEADS_;      // N*4 (uint-encoded max)
    float* ssum = m_f  + (size_t)N_ * HEADS_;      // N*4
    float* ew   = ssum + (size_t)N_ * HEADS_;      // E
    float* lpw  = ew   + (size_t)E_;               // N
    float* deg  = lpw  + (size_t)N_;               // N
    float* wmn  = deg  + (size_t)N_;               // ED
    float* cvec = wmn  + ED_;                      // HEADS
    float* bmn  = cvec + HEADS_;                   // 1
    size_t foff = (size_t)(bmn + 1 - ws);
    foff = (foff + 3u) & ~(size_t)3u;              // 16B align the bf16 tail
    __bf16* abuf = (__bf16*)(ws + foff);           // N*H halves
    __bf16* btbf = abuf + (size_t)N_ * H_;         // (4H)*H halves
    unsigned* mU = (unsigned*)m_f;

    const int TB = 256;
    auto blocks = [](long n, int tb) { return (int)((n + tb - 1) / tb); };

    // degree
    k_fill<<<blocks(N_, TB), TB, 0, stream>>>(deg, 0.f, N_);
    k_deg<<<blocks(E_, TB), TB, 0, stream>>>(dst0, deg);

    // node encoder: x = relu(NF @ enc_w + enc_b)
    k_cvt_bf16<<<blocks((long)N_ * ND_, TB), TB, 0, stream>>>(node_features, abuf, N_ * ND_);
    k_tr_bf16<<<blocks((long)ND_ * H_, TB), TB, 0, stream>>>(enc_w, btbf, ND_, H_);
    {
        dim3 g(H_ / BN, blocks(N_, BM));
        k_gemm_wmma<<<g, 256, 0, stream>>>(abuf, btbf, enc_b, x, N_, ND_, H_, 1);
    }

    for (int l = 0; l < L_; ++l) {
        k_prep_layer<<<1, 256, 0, stream>>>(
            edge_enc_w + (size_t)l * ED_ * H_, edge_enc_b + (size_t)l * H_,
            lin_edge_w + (size_t)l * HEADS_ * H_, att_edge + (size_t)l * HEADS_ * H_,
            wmn, cvec, bmn);
        k_fill<<<blocks(N_, TB), TB, 0, stream>>>(lpw, 0.f, N_);
        k_ew<<<blocks(E_, TB), TB, 0, stream>>>(edge_attr, dst0, wmn, bmn, ew, lpw);
        k_loopw_div<<<blocks(N_, TB), TB, 0, stream>>>(lpw, deg);

        // xp = x @ gat_w[l]   ([N,256] x [256,1024]) in bf16 WMMA
        k_cvt_bf16<<<blocks((long)N_ * H_, TB), TB, 0, stream>>>(x, abuf, N_ * H_);
        k_tr_bf16<<<blocks((long)H_ * HEADS_ * H_, TB), TB, 0, stream>>>(
            gat_w + (size_t)l * H_ * HEADS_ * H_, btbf, H_, HEADS_ * H_);
        {
            dim3 g((HEADS_ * H_) / BN, blocks(N_, BM));
            k_gemm_wmma<<<g, 256, 0, stream>>>(abuf, btbf, nullptr, xp,
                                               N_, H_, HEADS_ * H_, 0);
        }
        k_asd<<<blocks((long)N_ * HEADS_, 8), 256, 0, stream>>>(
            xp, att_src + (size_t)l * HEADS_ * H_, att_dst + (size_t)l * HEADS_ * H_,
            a_s, a_d);

        k_fill<<<blocks((long)N_ * HEADS_, TB), TB, 0, stream>>>(m_f, 0.f, N_ * HEADS_);
        k_fill<<<blocks((long)N_ * HEADS_, TB), TB, 0, stream>>>(ssum, 0.f, N_ * HEADS_);
        k_fill<<<blocks((long)N_ * H_, TB), TB, 0, stream>>>(hbuf, 0.f, N_ * H_);

        k_logit_max<<<blocks(ETOT_, TB), TB, 0, stream>>>(src0, dst0, ew, lpw,
                                                          a_s, a_d, cvec, mU);
        k_p_sum<<<blocks(ETOT_, TB), TB, 0, stream>>>(src0, dst0, ew, lpw,
                                                      a_s, a_d, cvec, mU, ssum);
        k_scatter<<<blocks(ETOT_, 8), 256, 0, stream>>>(src0, dst0, ew, lpw, a_s, a_d,
                                                        cvec, mU, ssum, xp, hbuf);
        k_post<<<blocks(N_, 8), 256, 0, stream>>>(x, hbuf, gat_b + (size_t)l * H_,
                                                  ln_g + (size_t)l * H_,
                                                  ln_b + (size_t)l * H_);
    }

    // outputs: [attrition, engagement, collaboration, anomaly] (N each) + emb (N*H)
    k_copy<<<blocks((long)N_ * H_, TB), TB, 0, stream>>>(x, out + 4 * N_, N_ * H_);
    k_heads<<<N_ * HEADS_, 128, 0, stream>>>(x, head_w1, head_b1, head_w2, head_b2, out);
}
